// GCN_24550033064494
// MI455X (gfx1250) — matrix operand order repaired and checked by reading, not compiled, not run on hardware
//
#include <hip/hip_runtime.h>
#include <hip/hip_bf16.h>

// ---------------------------------------------------------------------------
// Types for CDNA5 WMMA operands
// ---------------------------------------------------------------------------
typedef __attribute__((ext_vector_type(16))) __bf16 v16bf;
typedef __attribute__((ext_vector_type(8)))  __bf16 v8bf;
typedef __attribute__((ext_vector_type(8)))  float  v8f;
typedef __attribute__((ext_vector_type(2)))  float  v2f;

__device__ __forceinline__ v8f zero8() {
    v8f v;
#pragma unroll
    for (int i = 0; i < 8; ++i) v[i] = 0.0f;
    return v;
}

// ---------------------------------------------------------------------------
// Elementwise helpers
// ---------------------------------------------------------------------------
__global__ void cast_bf16_kernel(const float* __restrict__ in,
                                 __bf16* __restrict__ out, size_t n) {
    size_t i = (size_t)blockIdx.x * blockDim.x + threadIdx.x;
    if (i < n) out[i] = (__bf16)in[i];
}

// Tt[b,e,n] = (bf16) T[b,n,e]
__global__ void transpose_bf16_kernel(const float* __restrict__ T,
                                      __bf16* __restrict__ Tt,
                                      int B, int N, int E) {
    size_t i = (size_t)blockIdx.x * blockDim.x + threadIdx.x;
    size_t total = (size_t)B * E * N;
    if (i >= total) return;
    int n = (int)(i % N);
    int e = (int)((i / N) % E);
    int b = (int)(i / ((size_t)N * E));
    Tt[i] = (__bf16)T[((size_t)b * N + n) * E + e];
}

// w[b,r] = sum_k H[b,r,k] * p[k]
__global__ void gemv_w_kernel(const float* __restrict__ H,
                              const float* __restrict__ p,
                              float* __restrict__ w, int R, int F, int B) {
    int i = blockIdx.x * blockDim.x + threadIdx.x;
    if (i >= B * R) return;
    int b = i / R, r = i % R;
    const float* h = H + ((size_t)b * R + r) * F;
    float s = 0.0f;
    for (int k = 0; k < F; ++k) s += h[k] * p[k];
    w[i] = s;
}

// Split-scale: Sh = S * bf16(w), Sl = S * bf16(w - bf16(w));  S in {0,1} -> exact
__global__ void scale_split_kernel(const __bf16* __restrict__ S,
                                   const float* __restrict__ w,
                                   __bf16* __restrict__ Sh,
                                   __bf16* __restrict__ Sl,
                                   int R, int K, int B) {
    size_t i = (size_t)blockIdx.x * blockDim.x + threadIdx.x;
    size_t total = (size_t)B * R * K;
    if (i >= total) return;
    int k = (int)(i % K);
    int b = (int)(i / ((size_t)R * K));
    float wv = w[(size_t)b * K + k];
    __bf16 whi = (__bf16)wv;
    float  wlo = wv - (float)whi;
    float  s   = (float)S[i];
    Sh[i] = (__bf16)((float)whi * s);
    Sl[i] = (__bf16)(wlo * s);
}

// G[b,r,c] = sum_k H[b,r,k] * W[k,c] for c < f, else 0   (G padded to FP cols)
__global__ void gemm_g_kernel(const float* __restrict__ H,
                              const float* __restrict__ W,
                              float* __restrict__ G,
                              int R, int Kf, int f, int FP, int B) {
    size_t i = (size_t)blockIdx.x * blockDim.x + threadIdx.x;
    size_t total = (size_t)B * R * FP;
    if (i >= total) return;
    int c = (int)(i % FP);
    int r = (int)((i / FP) % R);
    int b = (int)(i / ((size_t)R * FP));
    float s = 0.0f;
    if (c < f) {
        const float* h = H + ((size_t)b * R + r) * Kf;
        for (int k = 0; k < Kf; ++k) s += h[k] * W[(size_t)k * f + c];
    }
    G[i] = s;
}

__global__ void relu_kernel(float* __restrict__ x, size_t n) {
    size_t i = (size_t)blockIdx.x * blockDim.x + threadIdx.x;
    if (i < n) x[i] = fmaxf(x[i], 0.0f);
}

__global__ void sigmoid_kernel(const float* __restrict__ h,
                               float* __restrict__ out, int n) {
    int i = blockIdx.x * blockDim.x + threadIdx.x;
    if (i < n) out[i] = 1.0f / (1.0f + expf(-h[i]));
}

// ---------------------------------------------------------------------------
// Fused masked-Gram + feature GEMM:
//   M = S diag(w) S^T  (split bf16 hi/lo WMMA, f32 accumulate)
//   A = (eye + (1-eye)*M) * adj
//   Out = A @ G + bias  (f32 WMMA 16x16x4), optional ReLU
// Block: 128 threads = 4 waves; grid = (R/16, B). Each wave owns JTILE=4
// consecutive j-tiles and strides by 16 tiles; cross-wave reduce via ds_add.
// ---------------------------------------------------------------------------
template <int NF16>
__global__ __launch_bounds__(128)
void gram_fused_kernel(const __bf16* __restrict__ S,    // [B,R,K]
                       const __bf16* __restrict__ Sh,   // [B,R,K]
                       const __bf16* __restrict__ Sl,   // [B,R,K]
                       const float*  __restrict__ adj,  // [B,R,R]
                       const float*  __restrict__ G,    // [B,R,NF16*16]
                       const float*  __restrict__ bias, // [f]
                       float*        __restrict__ Out,  // [B,R,f]
                       int R, int K, int f, int do_relu) {
    constexpr int FP = NF16 * 16;
    constexpr int JT = 4;
    const int b    = blockIdx.y;
    const int ib0  = blockIdx.x * 16;
    const int wave = threadIdx.x >> 5;
    const int lane = threadIdx.x & 31;
    const int l16  = lane & 15;
    const int hi   = lane >> 4;   // 0 for lanes 0-15, 1 for lanes 16-31

    __shared__ float ldsOut[16 * FP];
    __shared__ float ascr[4][16][17];   // per-wave masked A-tile (16x16, padded)

    for (int t = threadIdx.x; t < 16 * FP; t += blockDim.x) ldsOut[t] = 0.0f;
    __syncthreads();

    const size_t sb = (size_t)b * R * K;
    const __bf16* Sb  = S  + sb;
    const __bf16* Shb = Sh + sb;
    const __bf16* Slb = Sl + sb;
    const float*  Ab  = adj + (size_t)b * R * R;
    const float*  Gb  = G   + (size_t)b * R * FP;

    v8f acc[NF16];
#pragma unroll
    for (int t = 0; t < NF16; ++t) acc[t] = zero8();

    const int nj = R / 16;
    // A operand: row ib0+l16 of S; K-chunk split per half-wave (contiguous k = 16*hi + s, same order as B)
    const __bf16* aptr = Sb + (size_t)(ib0 + l16) * K;

    for (int jb = wave * JT; jb < nj; jb += 4 * JT) {
        v8f m[JT];
#pragma unroll
        for (int jj = 0; jj < JT; ++jj) m[jj] = zero8();

        for (int k0 = 0; k0 < K; k0 += 32) {
            union { v16bf v; v8bf h[2]; } a;
            a.h[0] = *(const v8bf*)(aptr + k0 + hi * 16);       // K 0..7 / 16..23
            a.h[1] = *(const v8bf*)(aptr + k0 + hi * 16 + 8);   // K 8..15 / 24..31
#pragma unroll
            for (int jj = 0; jj < JT; ++jj) {
                const size_t jrow = (size_t)((jb + jj) * 16 + l16) * K + k0 + hi * 16;
                v16bf bh = *(const v16bf*)(Shb + jrow);
                v16bf bl = *(const v16bf*)(Slb + jrow);
                m[jj] = __builtin_amdgcn_wmma_f32_16x16x32_bf16(
                            false, a.v, false, bh, (short)0, m[jj], false, false);
                m[jj] = __builtin_amdgcn_wmma_f32_16x16x32_bf16(
                            false, a.v, false, bl, (short)0, m[jj], false, false);
            }
        }

#pragma unroll
        for (int jj = 0; jj < JT; ++jj) {
            const int jb0 = (jb + jj) * 16;
            __syncthreads();   // protect per-wave ascr WAR across jj iterations
#pragma unroll
            for (int r = 0; r < 8; ++r) {
                int mi = ib0 + r + hi * 8;           // C layout: VGPR r -> row r / r+8
                int mj = jb0 + l16;                  // col = lane % 16
                float adjv = Ab[(size_t)mi * R + mj];
                float av = (mi == mj) ? adjv : m[jj][r] * adjv;
                ascr[wave][r + hi * 8][l16] = av;
            }
            __syncthreads();
            // A_tile(16x16) @ G[jb0:jb0+16, :FP] via f32 WMMA 16x16x4, K in 4 steps
#pragma unroll
            for (int ks = 0; ks < 4; ++ks) {
                v2f a2;
                a2.x = ascr[wave][l16][4 * ks + 2 * hi + 0];
                a2.y = ascr[wave][l16][4 * ks + 2 * hi + 1];
#pragma unroll
                for (int ft = 0; ft < NF16; ++ft) {
                    const float* gp = Gb + (size_t)(jb0 + 4 * ks) * FP + ft * 16 + l16;
                    v2f b2;
                    b2.x = gp[(size_t)(hi + 0) * FP];
                    b2.y = gp[(size_t)(hi + 2) * FP];
                    acc[ft] = __builtin_amdgcn_wmma_f32_16x16x4_f32(
                                  false, a2, false, b2, (short)0, acc[ft], false, false);
                }
            }
        }
    }

    // Cross-wave reduction via LDS float atomics
#pragma unroll
    for (int ft = 0; ft < NF16; ++ft)
#pragma unroll
        for (int r = 0; r < 8; ++r)
            atomicAdd(&ldsOut[(r + hi * 8) * FP + ft * 16 + l16], acc[ft][r]);
    __syncthreads();

    for (int t = threadIdx.x; t < 16 * FP; t += blockDim.x) {
        int row = t / FP, col = t % FP;
        if (col < f) {
            float v = ldsOut[t] + bias[col];
            if (do_relu) v = fmaxf(v, 0.0f);
            Out[((size_t)b * R + (ib0 + row)) * f + col] = v;
        }
    }
}

// ---------------------------------------------------------------------------
// Host orchestration
// ---------------------------------------------------------------------------
static void launch_gram(int NF16, const __bf16* S, const __bf16* Sh, const __bf16* Sl,
                        const float* adj, const float* G, const float* bias,
                        float* Out, int R, int K, int f, int relu, int B,
                        hipStream_t stream) {
    dim3 grid(R / 16, B), block(128);
    switch (NF16) {
    case 8: hipLaunchKernelGGL((gram_fused_kernel<8>), grid, block, 0, stream,
                               S, Sh, Sl, adj, G, bias, Out, R, K, f, relu); break;
    case 2: hipLaunchKernelGGL((gram_fused_kernel<2>), grid, block, 0, stream,
                               S, Sh, Sl, adj, G, bias, Out, R, K, f, relu); break;
    default: hipLaunchKernelGGL((gram_fused_kernel<1>), grid, block, 0, stream,
                               S, Sh, Sl, adj, G, bias, Out, R, K, f, relu); break;
    }
}

extern "C" void kernel_launch(void* const* d_in, const int* in_sizes, int n_in,
                              void* d_out, int out_size, void* d_ws, size_t ws_size,
                              hipStream_t stream) {
    (void)in_sizes; (void)n_in; (void)out_size; (void)ws_size;
    const int B = 2, N = 1024, E = 4096;

    const float* X     = (const float*)d_in[0];
    const float* Z     = (const float*)d_in[1];
    const float* adj_e = (const float*)d_in[2];
    const float* adj_v = (const float*)d_in[3];
    const float* T     = (const float*)d_in[4];

    // Workspace carve (256B aligned)
    size_t off = 0;
    auto carve = [&](size_t bytes) -> void* {
        void* p = (char*)d_ws + off;
        off += (bytes + 255) & ~(size_t)255;
        return p;
    };
    __bf16* Tbf  = (__bf16*)carve((size_t)B * N * E * 2);
    __bf16* Ttbf = (__bf16*)carve((size_t)B * E * N * 2);
    __bf16* Shb  = (__bf16*)carve((size_t)B * N * E * 2);
    __bf16* Slb  = (__bf16*)carve((size_t)B * N * E * 2);
    float*  Hv0  = (float*)carve((size_t)B * N * 512 * 4);
    float*  Hv1  = (float*)carve((size_t)B * N * 512 * 4);
    float*  He0  = (float*)carve((size_t)B * E * 512 * 4);
    float*  He1  = (float*)carve((size_t)B * E * 512 * 4);
    float*  Gbuf = (float*)carve((size_t)B * E * 128 * 4);
    float*  wbuf = (float*)carve((size_t)B * E * 4);

    // Init working copies & bf16 T / T^T
    hipMemcpyAsync(Hv0, X, (size_t)B * N * 512 * 4, hipMemcpyDeviceToDevice, stream);
    hipMemcpyAsync(He0, Z, (size_t)B * E * 512 * 4, hipMemcpyDeviceToDevice, stream);
    {
        size_t n = (size_t)B * N * E;
        cast_bf16_kernel<<<(n + 255) / 256, 256, 0, stream>>>(T, Tbf, n);
        transpose_bf16_kernel<<<(n + 255) / 256, 256, 0, stream>>>(T, Ttbf, B, N, E);
    }

    static const int IV[7]   = {512, 128, 128, 32, 32, 4, 4};
    static const int OV[7]   = {128, 128, 32, 32, 4, 4, 1};
    static const int IE[7]   = {512, 512, 128, 128, 32, 32, 4};
    static const int OE[7]   = {512, 128, 128, 32, 32, 4, 4};
    static const int NODE[7] = {1, 0, 1, 0, 1, 0, 1};

    float* hv_cur = Hv0; float* hv_alt = Hv1; int fv = 512;
    float* he_cur = He0; float* he_alt = He1; int fe = 512;

    for (int i = 0; i < 7; ++i) {
        const float* W  = (const float*)d_in[5 + 3 * i + 0];
        const float* p  = (const float*)d_in[5 + 3 * i + 1];
        const float* bb = (const float*)d_in[5 + 3 * i + 2];
        const int relu = (i < 6) ? 1 : 0;

        if (NODE[i]) {
            const int f = OV[i], FP = (f + 15) & ~15, NF16 = FP / 16;
            // w = He @ p  (length E per batch)
            gemv_w_kernel<<<(B * E + 255) / 256, 256, 0, stream>>>(he_cur, p, wbuf, E, fe, B);
            // Sh/Sl = T * diag(w_hi/lo)
            size_t n = (size_t)B * N * E;
            scale_split_kernel<<<(n + 255) / 256, 256, 0, stream>>>(Tbf, wbuf, Shb, Slb, N, E, B);
            // G = Hv @ W (padded to FP cols)
            size_t ng = (size_t)B * N * FP;
            gemm_g_kernel<<<(ng + 255) / 256, 256, 0, stream>>>(hv_cur, W, Gbuf, N, fv, f, FP, B);
            // Hv' = ((eye + (1-eye)*M) * adj_v) @ G + b
            launch_gram(NF16, Tbf, Shb, Slb, adj_v, Gbuf, bb, hv_alt, N, E, f, relu, B, stream);
            { float* t = hv_cur; hv_cur = hv_alt; hv_alt = t; }
            fv = f;
            if (i == 0) { // passthrough He = relu(Z); later passthroughs already >= 0
                size_t nr = (size_t)B * E * fe;
                relu_kernel<<<(nr + 255) / 256, 256, 0, stream>>>(he_cur, nr);
            }
        } else {
            const int f = OE[i], FP = (f + 15) & ~15, NF16 = FP / 16;
            gemv_w_kernel<<<(B * N + 255) / 256, 256, 0, stream>>>(hv_cur, p, wbuf, N, fv, B);
            size_t n = (size_t)B * E * N;
            scale_split_kernel<<<(n + 255) / 256, 256, 0, stream>>>(Ttbf, wbuf, Shb, Slb, E, N, B);
            size_t ng = (size_t)B * E * FP;
            gemm_g_kernel<<<(ng + 255) / 256, 256, 0, stream>>>(he_cur, W, Gbuf, E, fe, f, FP, B);
            launch_gram(NF16, Ttbf, Shb, Slb, adj_e, Gbuf, bb, he_alt, E, N, f, relu, B, stream);
            { float* t = he_cur; he_cur = he_alt; he_alt = t; }
            fe = f;
            (void)IE; (void)IV;
        }
    }

    // Output: sigmoid(Hv) [B,N,1]
    sigmoid_kernel<<<(B * N + 255) / 256, 256, 0, stream>>>(hv_cur, (float*)d_out, B * N);
}
